// GINLayer_24120536334770
// MI455X (gfx1250) — matrix operand order repaired and checked by reading, not compiled, not run on hardware
//
#include <hip/hip_runtime.h>
#include <hip/hip_bf16.h>

// ---------------------------------------------------------------------------
// GIN layer for MI455X (gfx1250, wave32):
//   agg[row] += x[col] (row!=col)  -> h0 = x + agg
//   h1 = relu(h0 @ W1 + b1)        -> WMMA f16 (f32 accum)
//   h2 = relu(h1 @ W2 + b2)        -> WMMA f16 (f32 accum) + fused BN stats
//   out = (h2-mean)*rsqrt(var+eps)*gamma + beta
// ---------------------------------------------------------------------------

typedef __attribute__((ext_vector_type(16))) _Float16 v16h;
typedef __attribute__((ext_vector_type(8)))  _Float16 v8h;
typedef __attribute__((ext_vector_type(4)))  _Float16 v4h;
typedef __attribute__((ext_vector_type(8)))  float    v8f;

#define GIN_N      100000
#define GIN_E      1600000
#define GIN_F      64
#define GIN_NTILES (GIN_N / 16)   // 6250 exact
#define GIN_EPS    1e-5f

// ---------------------------------------------------------------------------
// Kernel 0: agg = x (copy, float4) and zero the 128-float stats buffer.
// ---------------------------------------------------------------------------
__global__ __launch_bounds__(256) void k_init(const float* __restrict__ x,
                                              float* __restrict__ agg,
                                              float* __restrict__ gstats,
                                              int n4) {
    int i = blockIdx.x * 256 + threadIdx.x;
    if (i < n4) {
        ((float4*)agg)[i] = ((const float4*)x)[i];
    }
    if (i < 128) gstats[i] = 0.0f;
}

// ---------------------------------------------------------------------------
// Kernel 1: pack W [64x64 row-major, K x N] into per-lane WMMA B fragments.
// Fragment layout for v_wmma_f32_16x16x32_f16 (wave32):
//   lane L: n = t*16 + (L&15), kbase = s*32 + (L>>4)*8
//   v16h element e (VGPR j=e>>1, half e&1) holds W[kbase+(e&7)+(e>>3)*16][n].
// Stored as Wf[((t*2+s)*32 + L)*16 + e] so each lane loads 32 contiguous B.
// ---------------------------------------------------------------------------
__global__ __launch_bounds__(256) void k_pack_w(const float* __restrict__ W,
                                                _Float16* __restrict__ Wf) {
    int idx = blockIdx.x * 256 + threadIdx.x;   // 0..4095
    if (idx >= 4096) return;
    int e    = idx & 15;
    int lane = (idx >> 4) & 31;
    int ts   = idx >> 9;          // t*2 + s, 0..7
    int s    = ts & 1;
    int t    = ts >> 1;
    int k    = s * 32 + ((lane >> 4) * 8) + (e & 7) + ((e >> 3) * 16);
    int n    = t * 16 + (lane & 15);
    Wf[idx] = (_Float16)W[k * GIN_F + n];
}

// ---------------------------------------------------------------------------
// Kernel 2: edge scatter. 16 lanes per edge, float4 gather + 4 f32 atomics.
// edge_index laid out [2,E]: row = ei[e], col = ei[E+e].
// ---------------------------------------------------------------------------
__global__ __launch_bounds__(256) void k_scatter(const float* __restrict__ x,
                                                 const int* __restrict__ ei,
                                                 float* __restrict__ agg) {
    int tid = blockIdx.x * 256 + threadIdx.x;
    int e = tid >> 4;
    int t = tid & 15;
    if (e >= GIN_E) return;
    int r = ei[e];
    int c = ei[GIN_E + e];
    if (r == c) return;                               // remove_self_loops
    const float4 v = *(const float4*)(x + (size_t)c * GIN_F + t * 4);
    float* dst = agg + (size_t)r * GIN_F + t * 4;
    atomicAdd(dst + 0, v.x);
    atomicAdd(dst + 1, v.y);
    atomicAdd(dst + 2, v.z);
    atomicAdd(dst + 3, v.w);
}

// ---------------------------------------------------------------------------
// Kernel 3: cast h0 (f32) -> f16 for WMMA A operand.
// ---------------------------------------------------------------------------
__global__ __launch_bounds__(256) void k_cast16(const float* __restrict__ src,
                                                _Float16* __restrict__ dst,
                                                int n4) {
    int i = blockIdx.x * 256 + threadIdx.x;
    if (i >= n4) return;
    float4 v = ((const float4*)src)[i];
    v4h h;
    h[0] = (_Float16)v.x; h[1] = (_Float16)v.y;
    h[2] = (_Float16)v.z; h[3] = (_Float16)v.w;
    ((v4h*)dst)[i] = h;
}

// ---------------------------------------------------------------------------
// WMMA GEMM: [N,64] x [64,64] + bias, ReLU.
// One wave per 16-row tile; 8 waves per block; 4 n-tiles x 2 k-steps per wave.
// MODE 0: store f16 (feeds next GEMM).
// MODE 1: store f32 to out + fused BatchNorm sum/sumsq (LDS then global atomics).
// ---------------------------------------------------------------------------
template <int MODE>
__global__ __launch_bounds__(256) void k_gemm(const _Float16* __restrict__ A,
                                              const _Float16* __restrict__ Wf,
                                              const float* __restrict__ bias,
                                              void* __restrict__ outp,
                                              float* __restrict__ gstats) {
    __shared__ float ssum[GIN_F];
    __shared__ float ssq[GIN_F];
    if (MODE == 1) {
        if (threadIdx.x < GIN_F) {
            ssum[threadIdx.x] = 0.0f;
            ssq[threadIdx.x]  = 0.0f;
        }
        __syncthreads();
    }

    const int wave  = threadIdx.x >> 5;
    const int lane  = threadIdx.x & 31;
    const int tile  = blockIdx.x * 8 + wave;

    if (tile < GIN_NTILES) {          // wave-uniform: EXEC stays all-ones inside
        const int m     = lane & 15;
        const int khalf = lane >> 4;
        const int r0    = tile * 16;

        // A fragments: lane = row m, two contiguous 8-half chunks per k-step.
        const _Float16* arow = A + (size_t)(r0 + m) * GIN_F;
        v16h a0, a1;
        {
            v8h x0 = *(const v8h*)(arow + 0  + 8 * khalf);
            v8h x1 = *(const v8h*)(arow + 16 + 8 * khalf);
            v8h x2 = *(const v8h*)(arow + 32 + 8 * khalf);
            v8h x3 = *(const v8h*)(arow + 48 + 8 * khalf);
            #pragma unroll
            for (int i = 0; i < 8; ++i) {
                a0[i] = x0[i]; a0[i + 8] = x1[i];
                a1[i] = x2[i]; a1[i + 8] = x3[i];
            }
        }

        #pragma unroll
        for (int t = 0; t < 4; ++t) {
            const int nn = t * 16 + (lane & 15);
            v16h b0 = *(const v16h*)(Wf + (size_t)((t * 2 + 0) * 32 + lane) * 16);
            v16h b1 = *(const v16h*)(Wf + (size_t)((t * 2 + 1) * 32 + lane) * 16);

            const float bv = bias[nn];
            v8f acc;
            #pragma unroll
            for (int r = 0; r < 8; ++r) acc[r] = bv;

            acc = __builtin_amdgcn_wmma_f32_16x16x32_f16(
                false, a0, false, b0, (short)0, acc, false, false);
            acc = __builtin_amdgcn_wmma_f32_16x16x32_f16(
                false, a1, false, b1, (short)0, acc, false, false);

            const int rbase = r0 + 8 * khalf;   // C/D layout: VGPR r -> row rbase+r
            if (MODE == 0) {
                _Float16* o = (_Float16*)outp;
                #pragma unroll
                for (int r = 0; r < 8; ++r) {
                    float v = acc[r];
                    v = v > 0.0f ? v : 0.0f;
                    o[(size_t)(rbase + r) * GIN_F + nn] = (_Float16)v;
                }
            } else {
                float* o = (float*)outp;
                float ps = 0.0f, pss = 0.0f;
                #pragma unroll
                for (int r = 0; r < 8; ++r) {
                    float v = acc[r];
                    v = v > 0.0f ? v : 0.0f;
                    o[(size_t)(rbase + r) * GIN_F + nn] = v;
                    ps  += v;
                    pss += v * v;
                }
                atomicAdd(&ssum[nn], ps);   // ds_add_f32
                atomicAdd(&ssq[nn],  pss);
            }
        }
    }

    if (MODE == 1) {
        __syncthreads();
        if (threadIdx.x < GIN_F) {
            atomicAdd(&gstats[threadIdx.x],         ssum[threadIdx.x]);
            atomicAdd(&gstats[GIN_F + threadIdx.x], ssq[threadIdx.x]);
        }
    }
}

// ---------------------------------------------------------------------------
// Kernel 6: finalize BN stats -> per-feature scale/shift.
// var = E[h^2] - E[h]^2 (biased), matching the reference.
// ---------------------------------------------------------------------------
__global__ void k_stats(const float* __restrict__ gstats,
                        const float* __restrict__ gamma,
                        const float* __restrict__ beta,
                        float* __restrict__ scsh) {
    int f = threadIdx.x;
    if (f >= GIN_F) return;
    const float inv_n = 1.0f / (float)GIN_N;
    float mean = gstats[f] * inv_n;
    float var  = gstats[GIN_F + f] * inv_n - mean * mean;
    float sc   = gamma[f] * rsqrtf(var + GIN_EPS);
    scsh[f]         = sc;
    scsh[GIN_F + f] = beta[f] - mean * sc;
}

// ---------------------------------------------------------------------------
// Kernel 7: in-place BN apply on d_out (float4).
// ---------------------------------------------------------------------------
__global__ __launch_bounds__(256) void k_bn(float* __restrict__ h2,
                                            const float* __restrict__ scsh,
                                            int n4) {
    int i = blockIdx.x * 256 + threadIdx.x;
    if (i >= n4) return;
    float4 v = ((float4*)h2)[i];
    int fb = (i & 15) * 4;                    // 64 feats / 4 per float4
    v.x = v.x * scsh[fb + 0] + scsh[GIN_F + fb + 0];
    v.y = v.y * scsh[fb + 1] + scsh[GIN_F + fb + 1];
    v.z = v.z * scsh[fb + 2] + scsh[GIN_F + fb + 2];
    v.w = v.w * scsh[fb + 3] + scsh[GIN_F + fb + 3];
    ((float4*)h2)[i] = v;
}

// ---------------------------------------------------------------------------
// Launch. Inputs: x, edge_index, W1, b1, W2, b2, gamma, beta.
// ---------------------------------------------------------------------------
extern "C" void kernel_launch(void* const* d_in, const int* in_sizes, int n_in,
                              void* d_out, int out_size, void* d_ws, size_t ws_size,
                              hipStream_t stream) {
    (void)in_sizes; (void)n_in; (void)out_size; (void)ws_size;

    const float* x     = (const float*)d_in[0];
    const int*   ei    = (const int*)d_in[1];
    const float* W1    = (const float*)d_in[2];
    const float* b1    = (const float*)d_in[3];
    const float* W2    = (const float*)d_in[4];
    const float* b2    = (const float*)d_in[5];
    const float* gamma = (const float*)d_in[6];
    const float* beta  = (const float*)d_in[7];
    float*       out   = (float*)d_out;

    // Workspace layout (256-B aligned offsets), ~51.3 MB total.
    char* ws = (char*)d_ws;
    float*    agg    = (float*)(ws + 0);                 // N*64*4   = 25,600,000
    _Float16* h0h    = (_Float16*)(ws + 25600000);       // N*64*2   = 12,800,000
    _Float16* h1h    = (_Float16*)(ws + 38400000);       // N*64*2   = 12,800,000
    _Float16* W1f    = (_Float16*)(ws + 51200000);       // 4096*2   = 8192
    _Float16* W2f    = (_Float16*)(ws + 51208192);       // 8192
    float*    gstats = (float*)(ws + 51216384);          // 128 floats (sum|sumsq)
    float*    scsh   = (float*)(ws + 51216896);          // 128 floats (scale|shift)

    const int n4 = GIN_N * GIN_F / 4;                    // 1,600,000 float4s
    const int elem_blocks = (n4 + 255) / 256;            // 6250
    const int gemm_blocks = (GIN_NTILES + 7) / 8;        // 782
    const int edge_blocks = (GIN_E * 16) / 256;          // 100,000

    k_init<<<elem_blocks, 256, 0, stream>>>(x, agg, gstats, n4);
    k_pack_w<<<16, 256, 0, stream>>>(W1, W1f);
    k_pack_w<<<16, 256, 0, stream>>>(W2, W2f);
    k_scatter<<<edge_blocks, 256, 0, stream>>>(x, ei, agg);
    k_cast16<<<elem_blocks, 256, 0, stream>>>(agg, h0h, n4);
    k_gemm<0><<<gemm_blocks, 256, 0, stream>>>(h0h, W1f, b1, (void*)h1h, nullptr);
    k_gemm<1><<<gemm_blocks, 256, 0, stream>>>(h1h, W2f, b2, (void*)out, gstats);
    k_stats<<<1, 64, 0, stream>>>(gstats, gamma, beta, scsh);
    k_bn<<<elem_blocks, 256, 0, stream>>>(out, scsh, n4);
}
